// NeuroSAT_83949430767707
// MI455X (gfx1250) — matrix operand order repaired and checked by reading, not compiled, not run on hardware
//
#include <hip/hip_runtime.h>
#include <hip/hip_bf16.h>

// ---------------------------------------------------------------------------
// NeuroSAT forward for MI455X (gfx1250, wave32, WMMA + Tensor Data Mover).
// GEMMs: K=128 bf16 WMMA (v_wmma_f32_16x16x32_bf16, f32 accumulate).
// Tiles staged into LDS by TDM (tensor_load_to_lds, TENSORcnt), with a
// cooperative-copy fallback when the builtin is unavailable.
// Sparse adjacency handled with global_atomic_add_f32 scatter/gather.
// ---------------------------------------------------------------------------

typedef __attribute__((ext_vector_type(16))) __bf16 v16bf;
typedef __attribute__((ext_vector_type(8)))  __bf16 v8bf;
typedef __attribute__((ext_vector_type(8)))  float  v8f;
typedef unsigned int u32x4 __attribute__((ext_vector_type(4)));
typedef int          i32x4 __attribute__((ext_vector_type(4)));
typedef int          i32x8 __attribute__((ext_vector_type(8)));

#define N_VARS    3000
#define N_LITS    6000
#define N_CLAUSES 12000
#define N_CELLS   48000
#define N_PROBS   50
#define DIM       128
#define N_ROUNDS  16

#define FLAG_RELU 1
#define FLAG_ACC  2

#if defined(__has_builtin)
#if __has_builtin(__builtin_amdgcn_tensor_load_to_lds) && \
    __has_builtin(__builtin_amdgcn_s_wait_tensorcnt)
#define HAVE_TDM 1
#endif
#endif
#ifndef HAVE_TDM
#define HAVE_TDM 0
#endif

__device__ __forceinline__ float sigmoidf_(float x) {
    return 1.0f / (1.0f + __expf(-x));
}

#if HAVE_TDM
// Issue a 2-D TDM tile load: 128 rows x 128 bf16 elements into LDS.
// D# per CDNA5 ISA ch.8: group0 = {flags, lds_addr, global_addr, type=2},
// group1 = {data_size | tensor dims | tile dims | dim0 stride};
// groups 2/3 unused (2-D tensor).  Rows beyond tensor_d1 read as zero
// (hardware OOB) -> free M-tail padding.
__device__ __forceinline__ void tdm_load_2d_bf16(unsigned int lds_off,
                                                 const __bf16* gptr,
                                                 int tensor_d1,
                                                 unsigned int d0_stride)
{
    const unsigned long long ga = (unsigned long long)(uintptr_t)gptr;
    const unsigned int td0 = 128u, tile_d0 = 128u, tile_d1 = 128u;
    u32x4 g0;
    g0[0] = 1u;                                   // count=1, user descriptor
    g0[1] = lds_off;                              // lds_addr (bytes)
    g0[2] = (unsigned int)ga;                     // global_addr[31:0]
    g0[3] = (unsigned int)((ga >> 32) & 0x1FFFFFFu) | (2u << 30); // [56:32]|type=2
    const unsigned int td1 = (unsigned int)(tensor_d1 < 0 ? 0 : tensor_d1);
    i32x8 g1;
    g1[0] = (int)(1u << 16);                      // wg_mask=0, data_size=1 (2B)
    g1[1] = (int)((td0 & 0xFFFFu) << 16);         // abar=0 | tensor_dim0[15:0]
    g1[2] = (int)((td0 >> 16) | ((td1 & 0xFFFFu) << 16));
    g1[3] = (int)((td1 >> 16) | (tile_d0 << 16)); // tensor_dim1[31:16] | tile_dim0
    g1[4] = (int)(tile_d1 & 0xFFFFu);             // tile_dim1 | tile_dim2=0
    g1[5] = (int)d0_stride;                       // tensor_dim0_stride[31:0]
    g1[6] = 0;                                    // stride hi | dim1_stride lo
    g1[7] = 0;
    const i32x4 z4 = {0, 0, 0, 0};
#if defined(__clang_major__) && (__clang_major__ <= 22)
    // ROCm 7.2 / clang-22: 5-arg form (g0, g1, g2, g3, cpol)
    __builtin_amdgcn_tensor_load_to_lds(g0, g1, z4, z4, 0);
#else
    // amdgpu-toolchain / clang-23+: 6-arg form (g0, g1, g2, g3, g4, cpol)
    const i32x8 z8 = {0, 0, 0, 0, 0, 0, 0, 0};
    __builtin_amdgcn_tensor_load_to_lds(g0, g1, z4, z4, z8, 0);
#endif
}
#endif

// ---------------------------------------------------------------------------
// C[m, n0+n] (+)= A[m, 0:128] @ W[n0+n, k0:k0+128]^T   (+bias)(+relu)
// A: bf16 [M,128] row-major.  W: bf16 row-major, row stride ldb.
// Block: 256 threads (8 waves), 128x128 output tile.  grid.y = N/128.
// ---------------------------------------------------------------------------
__global__ __launch_bounds__(256)
void k_gemm_bf16(const __bf16* __restrict__ A, int M,
                 const __bf16* __restrict__ W, int ldb, int k0,
                 const float* __restrict__ bias,
                 float* __restrict__ C, int ldc, int flags)
{
    __shared__ __align__(32) __bf16 sA[128 * 128];   // [m][k], 32 KB
    __shared__ __align__(32) __bf16 sB[128 * 128];   // [n][k], 32 KB

    const int tid = threadIdx.x;
    const int n0  = blockIdx.y * 128;
    const int m0  = blockIdx.x * 128;

#if HAVE_TDM
    if (tid < 32) {                                  // wave 0 drives the DMAs
        const unsigned int offA = (unsigned int)(uintptr_t)&sA[0];
        const unsigned int offB = (unsigned int)(uintptr_t)&sB[0];
        tdm_load_2d_bf16(offA, A + (size_t)m0 * DIM, M - m0, DIM);
        tdm_load_2d_bf16(offB, W + (size_t)n0 * ldb + k0, 128, (unsigned int)ldb);
        __builtin_amdgcn_s_wait_tensorcnt(0);
    }
    __syncthreads();
#else
    for (int i = tid; i < 128 * 128; i += 256) {
        const int n = i >> 7, k = i & 127;
        sB[i] = W[(size_t)(n0 + n) * ldb + k0 + k];
    }
    for (int i = tid; i < 128 * 128; i += 256) {
        const int r = i >> 7, k = i & 127;
        const int row = m0 + r;
        sA[i] = (row < M) ? A[(size_t)row * DIM + k] : (__bf16)0.0f;
    }
    __syncthreads();
#endif

    const int w    = tid >> 5;        // wave id: owns rows [w*16, w*16+16)
    const int lane = tid & 31;
    const int hw   = lane >> 4;       // half-wave
    const int lr   = lane & 15;

    v8f acc[8] = {};                  // 8 N-tiles x 16x16 f32

    const int arow = (w * 16 + lr) * 128;
    const int brow = lr * 128 + 16 * hw;
    #pragma unroll
    for (int ks = 0; ks < 4; ++ks) {  // K = 4 x 32
        const int kb = ks * 32;
        // A fragment 16x32: lanes 0-15 K=0-15; lanes 16-31 K=16-31 (matches B)
        v8bf alo = *(const v8bf*)&sA[arow + kb + 16 * hw];
        v8bf ahi = *(const v8bf*)&sA[arow + kb + 16 * hw + 8];
        v16bf af = __builtin_shufflevector(alo, ahi,
                     0, 1, 2, 3, 4, 5, 6, 7, 8, 9, 10, 11, 12, 13, 14, 15);
        // Preload all 8 B fragments for this K-step, then issue WMMAs
        // back-to-back (single DS wait, dense XDL issue).
        v16bf bf[8];
        #pragma unroll
        for (int t = 0; t < 8; ++t)
            bf[t] = *(const v16bf*)&sB[t * 2048 + brow + kb];
        #pragma unroll
        for (int t = 0; t < 8; ++t)
            acc[t] = __builtin_amdgcn_wmma_f32_16x16x32_bf16(
                         false, af, false, bf[t], (short)0, acc[t], false, false);
    }

    // Epilogue: D layout VGPR r -> m = r + 8*(lane/16), n = lane%16.
    #pragma unroll
    for (int t = 0; t < 8; ++t) {
        #pragma unroll
        for (int r = 0; r < 8; ++r) {
            const int m = m0 + w * 16 + r + 8 * hw;
            if (m >= M) continue;
            const int n = t * 16 + lr;
            float v = acc[t][r];
            if (bias) v += bias[n0 + n];
            const size_t idx = (size_t)m * ldc + n0 + n;
            if (flags & FLAG_ACC) v += C[idx];
            if (flags & FLAG_RELU) v = fmaxf(v, 0.0f);
            C[idx] = v;
        }
    }
}

// ---------------------------------------------------------------------------
// Elementwise / sparse kernels
// ---------------------------------------------------------------------------
__global__ void k_init(float* __restrict__ hL, float* __restrict__ cL,
                       float* __restrict__ hC, float* __restrict__ cC,
                       const float* __restrict__ Lw, const float* __restrict__ Lb,
                       const float* __restrict__ Cw, const float* __restrict__ Cb)
{
    const int i = blockIdx.x * 256 + threadIdx.x;
    if (i >= N_CLAUSES * DIM) return;
    const int d = i & 127;
    hC[i] = Cw[d] + Cb[d];
    cC[i] = 0.0f;
    if (i < N_LITS * DIM) { hL[i] = Lw[d] + Lb[d]; cL[i] = 0.0f; }
}

__global__ void k_zero(float* __restrict__ p, int n)
{
    const int i = blockIdx.x * 256 + threadIdx.x;
    if (i < n) p[i] = 0.0f;
}

// f32 -> bf16 (TDM-ready A tiles / weight conversion)
__global__ void k_cvt(const float* __restrict__ in, __bf16* __restrict__ out, int n)
{
    const int i = blockIdx.x * 256 + threadIdx.x;
    if (i < n) out[i] = (__bf16)in[i];
}

// flipped(hL) = concat([hL[n_vars:], hL[:n_vars]]) as bf16
__global__ void k_cvt_flip(const float* __restrict__ hL, __bf16* __restrict__ out)
{
    const int i = blockIdx.x * 256 + threadIdx.x;
    if (i >= N_LITS * DIM) return;
    const int m = i >> 7, d = i & 127;
    int src = m + N_VARS; if (src >= N_LITS) src -= N_LITS;
    out[i] = (__bf16)hL[(size_t)src * DIM + d];
}

// dst[didx[c], :] += src[sidx[c], :]  for each of the 48000 cells.
__global__ void k_scatter_add(float* __restrict__ dst, const int* __restrict__ didx,
                              const float* __restrict__ src, const int* __restrict__ sidx)
{
    const int i = blockIdx.x * 256 + threadIdx.x;   // N_CELLS * DIM
    if (i >= N_CELLS * DIM) return;
    const int c = i >> 7, d = i & 127;
    const float v = src[(size_t)sidx[c] * DIM + d];
    __hip_atomic_fetch_add(&dst[(size_t)didx[c] * DIM + d], v,
                           __ATOMIC_RELAXED, __HIP_MEMORY_SCOPE_AGENT);
}

// PyTorch LSTM cell, gate order i,f,g,o laid out along N of gates[M,512].
__global__ void k_lstm(const float* __restrict__ g, float* __restrict__ h,
                       float* __restrict__ c, int M)
{
    const int i = blockIdx.x * 256 + threadIdx.x;
    if (i >= M * DIM) return;
    const int m = i >> 7, d = i & 127;
    const float* gr = g + (size_t)m * 512;
    const float ig = sigmoidf_(gr[d]);
    const float fg = sigmoidf_(gr[128 + d]);
    const float gg = tanhf(gr[256 + d]);
    const float og = sigmoidf_(gr[384 + d]);
    const float cn = fg * c[i] + ig * gg;
    c[i] = cn;
    h[i] = og * tanhf(cn);
}

// vote[v] = V[v,:] . w3 + b3    (final 128 -> 1 layer)
__global__ void k_vote(const float* __restrict__ V, const float* __restrict__ w3,
                       const float* __restrict__ b3, float* __restrict__ vote)
{
    const int v = blockIdx.x * 256 + threadIdx.x;
    if (v >= N_LITS) return;
    float s = b3[0];
    #pragma unroll 4
    for (int k = 0; k < DIM; ++k) s += V[(size_t)v * DIM + k] * w3[k];
    vote[v] = s;
}

// out[p] = mean over 60 vars x 2 polarities
__global__ void k_mean(const float* __restrict__ vote, float* __restrict__ out)
{
    const int p = threadIdx.x;
    if (p >= N_PROBS) return;
    const int vpp = N_VARS / N_PROBS;   // 60
    float s = 0.0f;
    const int base = p * vpp;
    for (int j = 0; j < vpp; ++j)
        s += vote[base + j] + vote[N_VARS + base + j];
    out[p] = s / (float)(2 * vpp);
}

// ---------------------------------------------------------------------------
// Host-side orchestration
// ---------------------------------------------------------------------------
extern "C" void kernel_launch(void* const* d_in, const int* in_sizes, int n_in,
                              void* d_out, int out_size, void* d_ws, size_t ws_size,
                              hipStream_t stream)
{
    (void)in_sizes; (void)n_in; (void)out_size; (void)ws_size;

    // Flattened setup_inputs() order
    const int*   lit_idx    = (const int*)d_in[0];
    const int*   clause_idx = (const int*)d_in[1];
    const float* L_init_w = (const float*)d_in[4];
    const float* L_init_b = (const float*)d_in[5];
    const float* C_init_w = (const float*)d_in[6];
    const float* C_init_b = (const float*)d_in[7];
    const float* Lm_w1 = (const float*)d_in[8],  *Lm_b1 = (const float*)d_in[9];
    const float* Lm_w2 = (const float*)d_in[10], *Lm_b2 = (const float*)d_in[11];
    const float* Lm_w3 = (const float*)d_in[12], *Lm_b3 = (const float*)d_in[13];
    const float* Cm_w1 = (const float*)d_in[14], *Cm_b1 = (const float*)d_in[15];
    const float* Cm_w2 = (const float*)d_in[16], *Cm_b2 = (const float*)d_in[17];
    const float* Cm_w3 = (const float*)d_in[18], *Cm_b3 = (const float*)d_in[19];
    const float* Lu_wih = (const float*)d_in[20], *Lu_bih = (const float*)d_in[21];
    const float* Lu_whh = (const float*)d_in[22], *Lu_bhh = (const float*)d_in[23];
    const float* Cu_wih = (const float*)d_in[24], *Cu_bih = (const float*)d_in[25];
    const float* Cu_whh = (const float*)d_in[26], *Cu_bhh = (const float*)d_in[27];
    const float* Lv_w1 = (const float*)d_in[28], *Lv_b1 = (const float*)d_in[29];
    const float* Lv_w2 = (const float*)d_in[30], *Lv_b2 = (const float*)d_in[31];
    const float* Lv_w3 = (const float*)d_in[32], *Lv_b3 = (const float*)d_in[33];

    float* out = (float*)d_out;

    // Workspace layout: f32 region, then bf16 region (~59 MB total)
    float* ws   = (float*)d_ws;
    float* hL   = ws;                              // 6000*128
    float* cL   = hL + (size_t)N_LITS * DIM;
    float* hC   = cL + (size_t)N_LITS * DIM;       // 12000*128
    float* cC   = hC + (size_t)N_CLAUSES * DIM;
    float* t0   = cC + (size_t)N_CLAUSES * DIM;    // scratch 12000*128
    float* t1   = t0 + (size_t)N_CLAUSES * DIM;    // scratch 12000*128
    float* G    = t1 + (size_t)N_CLAUSES * DIM;    // gates  12000*512 (shared)
    float* vote = G  + (size_t)N_CLAUSES * 4 * DIM;// 6000 (padded to 8192)

    __bf16* bb      = (__bf16*)(vote + 8192);      // bf16 region (256B aligned)
    __bf16* Lm_w1b  = bb;                 __bf16* Lm_w2b = Lm_w1b + 16384;
    __bf16* Lm_w3b  = Lm_w2b + 16384;     __bf16* Cm_w1b = Lm_w3b + 16384;
    __bf16* Cm_w2b  = Cm_w1b + 16384;     __bf16* Cm_w3b = Cm_w2b + 16384;
    __bf16* Lv_w1b  = Cm_w3b + 16384;     __bf16* Lv_w2b = Lv_w1b + 16384;
    __bf16* Cu_wihb = Lv_w2b + 16384;     __bf16* Cu_whhb = Cu_wihb + 65536;
    __bf16* Lu_whhb = Cu_whhb + 65536;    __bf16* Lu_wihb = Lu_whhb + 65536;
    __bf16* Ab      = Lu_wihb + 131072;   // A staging, 12000*128 bf16

    auto cvt = [&](const float* src, __bf16* dst, int n) {
        k_cvt<<<(n + 255) / 256, 256, 0, stream>>>(src, dst, n);
    };
    auto gemm = [&](const __bf16* A, int M, const __bf16* W, int ldb, int k0,
                    const float* bias, float* C, int ldc, int ntiles, int flags) {
        dim3 grid((M + 127) / 128, ntiles);
        k_gemm_bf16<<<grid, 256, 0, stream>>>(A, M, W, ldb, k0, bias, C, ldc, flags);
    };

    const int gC = (N_CLAUSES * DIM + 255) / 256;
    const int gL = (N_LITS * DIM + 255) / 256;
    const int gS = (N_CELLS * DIM + 255) / 256;
    const int eC = N_CLAUSES * DIM, eL = N_LITS * DIM;

    // One-time weight conversion to bf16 (TDM-ready, n-major == W row-major)
    cvt(Lm_w1, Lm_w1b, 16384);  cvt(Lm_w2, Lm_w2b, 16384);  cvt(Lm_w3, Lm_w3b, 16384);
    cvt(Cm_w1, Cm_w1b, 16384);  cvt(Cm_w2, Cm_w2b, 16384);  cvt(Cm_w3, Cm_w3b, 16384);
    cvt(Lv_w1, Lv_w1b, 16384);  cvt(Lv_w2, Lv_w2b, 16384);
    cvt(Cu_wih, Cu_wihb, 65536); cvt(Cu_whh, Cu_whhb, 65536);
    cvt(Lu_whh, Lu_whhb, 65536); cvt(Lu_wih, Lu_wihb, 131072);

    k_init<<<gC, 256, 0, stream>>>(hL, cL, hC, cC,
                                   L_init_w, L_init_b, C_init_w, C_init_b);

    for (int r = 0; r < N_ROUNDS; ++r) {
        // L_msg MLP: t0 = mlp(hL)
        cvt(hL, Ab, eL);  gemm(Ab, N_LITS, Lm_w1b, 128, 0, Lm_b1, t0, 128, 1, FLAG_RELU);
        cvt(t0, Ab, eL);  gemm(Ab, N_LITS, Lm_w2b, 128, 0, Lm_b2, t1, 128, 1, FLAG_RELU);
        cvt(t1, Ab, eL);  gemm(Ab, N_LITS, Lm_w3b, 128, 0, Lm_b3, t0, 128, 1, 0);

        // LC_msg = A^T @ t0  (sparse scatter): t1[clause] += t0[lit]
        k_zero<<<gC, 256, 0, stream>>>(t1, eC);
        k_scatter_add<<<gS, 256, 0, stream>>>(t1, clause_idx, t0, lit_idx);

        // Clause LSTM: G = LC @ wih^T + bih ; G += hC @ whh^T + bhh
        cvt(t1, Ab, eC);  gemm(Ab, N_CLAUSES, Cu_wihb, 128, 0, Cu_bih, G, 512, 4, 0);
        cvt(hC, Ab, eC);  gemm(Ab, N_CLAUSES, Cu_whhb, 128, 0, Cu_bhh, G, 512, 4, FLAG_ACC);
        k_lstm<<<gC, 256, 0, stream>>>(G, hC, cC, N_CLAUSES);

        // C_msg MLP: t0 = mlp(hC)
        cvt(hC, Ab, eC);  gemm(Ab, N_CLAUSES, Cm_w1b, 128, 0, Cm_b1, t0, 128, 1, FLAG_RELU);
        cvt(t0, Ab, eC);  gemm(Ab, N_CLAUSES, Cm_w2b, 128, 0, Cm_b2, t1, 128, 1, FLAG_RELU);
        cvt(t1, Ab, eC);  gemm(Ab, N_CLAUSES, Cm_w3b, 128, 0, Cm_b3, t0, 128, 1, 0);

        // CL_msg = A @ t0 (sparse gather): t1[lit] += t0[clause]
        k_zero<<<gL, 256, 0, stream>>>(t1, eL);
        k_scatter_add<<<gS, 256, 0, stream>>>(t1, lit_idx, t0, clause_idx);

        // Literal LSTM: x = [CL_msg, flipped(hL)] -> two K=128 passes over wih
        cvt(t1, Ab, eL);  gemm(Ab, N_LITS, Lu_wihb, 256, 0,   Lu_bih,  G, 512, 4, 0);
        k_cvt_flip<<<gL, 256, 0, stream>>>(hL, Ab);
                          gemm(Ab, N_LITS, Lu_wihb, 256, 128, nullptr, G, 512, 4, FLAG_ACC);
        cvt(hL, Ab, eL);  gemm(Ab, N_LITS, Lu_whhb, 128, 0,   Lu_bhh,  G, 512, 4, FLAG_ACC);
        k_lstm<<<gL, 256, 0, stream>>>(G, hL, cL, N_LITS);
    }

    // Vote MLP + per-problem mean
    cvt(hL, Ab, eL);  gemm(Ab, N_LITS, Lv_w1b, 128, 0, Lv_b1, t0, 128, 1, FLAG_RELU);
    cvt(t0, Ab, eL);  gemm(Ab, N_LITS, Lv_w2b, 128, 0, Lv_b2, t1, 128, 1, FLAG_RELU);
    k_vote<<<(N_LITS + 255) / 256, 256, 0, stream>>>(t1, Lv_w3, Lv_b3, vote);
    k_mean<<<1, 64, 0, stream>>>(vote, out);
}
